// GraphDecoder_88923002896567
// MI455X (gfx1250) — compile-verified, matrix operand order checked by reading, not run on hardware
//
#include <hip/hip_runtime.h>
#include <hip/hip_bf16.h>

#define L_NUM   6
#define DM      512
#define NH      8
#define DFF     2048
#define SQ      64
#define NIMG    512
#define NOBJ    10240
#define DH      64
#define FLAT_ELEMS ((size_t)NOBJ * DM)              // 5,242,880
#define ATT_PER_HEAD ((size_t)SQ * SQ)              // 4096

typedef _Float16 v16h __attribute__((ext_vector_type(16)));
typedef _Float16 v8h  __attribute__((ext_vector_type(8)));
typedef float    v8f  __attribute__((ext_vector_type(8)));

typedef unsigned int u32x4 __attribute__((ext_vector_type(4)));
typedef int          i32x4 __attribute__((ext_vector_type(4)));
typedef int          i32x8 __attribute__((ext_vector_type(8)));

#if defined(__has_builtin)
# if __has_builtin(__builtin_amdgcn_tensor_store_from_lds)
#  define HAVE_TDM 1
# endif
#endif

// ---------------------------------------------------------------------------
// Fragment load: 16x32 f16 operand per CDNA5 WMMA lane layout.
// lanes 0-15 : row M=lane,    K = {k0+0..7, k0+16..23}
// lanes 16-31: row M=lane-16, K = {k0+8..15, k0+24..31}
// "base" is row-major [rows][ld] f16; used for A and for B^T ([N][K]).
// Both halves are aligned 16B -> ds_load_b128 / global_load_b128.
// ---------------------------------------------------------------------------
__device__ __forceinline__ v16h load_frag(const _Float16* __restrict__ base,
                                          int ld, int r0, int k0) {
    const int lane = threadIdx.x & 31;
    const int m    = lane & 15;
    const int kb   = (lane & 16) ? 8 : 0;
    const _Float16* p = base + (size_t)(r0 + m) * ld + (k0 + kb);
    v8h lo = *(const v8h*)(p);
    v8h hi = *(const v8h*)(p + 16);
    return __builtin_shufflevector(lo, hi,
        0,1,2,3,4,5,6,7,8,9,10,11,12,13,14,15);
}

__device__ __forceinline__ void epilogue_tile(
    v8f acc, int mt, int nt, int rowb, int colw,
    float* cf32, _Float16* cf16, int ldc,
    const float* __restrict__ bias, int bias0,
    float scale, bool relu, bool storeT)
{
    const float bvv = bias ? bias[bias0 + nt * 16 + colw] : 0.0f;
#pragma unroll
    for (int r = 0; r < 8; ++r) {
        float v = acc[r] * scale + bvv;
        if (relu) v = fmaxf(v, 0.0f);
        const int row = mt * 16 + rowb + r;
        const int col = nt * 16 + colw;
        if (cf32) cf32[(size_t)row * ldc + col] = v;
        if (cf16) {
            if (storeT) cf16[(size_t)col * ldc + row] = (_Float16)v;
            else        cf16[(size_t)row * ldc + col] = (_Float16)v;
        }
    }
}

// ---------------------------------------------------------------------------
// M=64 WMMA GEMM: C(64 x Ntiles*16) = A(64 x Kblocks*32) * B + [C]
//   A  : LDS f16 row-major (lda);  BT : f16 B transposed [N][K] (ldb)
// Each wave processes a PAIR of adjacent column tiles (same row tile,
// shared A fragment) with two independent accumulators, so consecutive
// v_wmma ops have no RAW dependence and the WMMA->WMMA hazard slots are
// filled with useful work instead of v_nops. Requires Ntiles even (4 or 32).
// ---------------------------------------------------------------------------
__device__ __forceinline__ void wmma_gemm(
    const _Float16* __restrict__ A, int lda,
    const _Float16* __restrict__ BT, int ldb,
    int Ntiles, int Kblocks,
    float* cf32, _Float16* cf16, int ldc,
    const float* __restrict__ bias, int bias0,
    float scale, bool relu, bool accum, bool storeT)
{
    const int wave = threadIdx.x >> 5;
    const int lane = threadIdx.x & 31;
    const int colw = lane & 15;
    const int rowb = (lane & 16) ? 8 : 0;

    for (int t = 2 * wave; t < 4 * Ntiles; t += 16) {
        const int mt  = t / Ntiles;         // t even & Ntiles even -> nt0+1 < Ntiles
        const int nt0 = t % Ntiles;
        const int nt1 = nt0 + 1;

        v8f acc0 = {}, acc1 = {};
        if (accum) {
#pragma unroll
            for (int r = 0; r < 8; ++r) {
                acc0[r] = cf32[(size_t)(mt * 16 + rowb + r) * ldc + nt0 * 16 + colw];
                acc1[r] = cf32[(size_t)(mt * 16 + rowb + r) * ldc + nt1 * 16 + colw];
            }
        }
        for (int kb = 0; kb < Kblocks; ++kb) {
            v16h a  = load_frag(A,  lda, mt * 16,  kb * 32);   // shared A
            v16h b0 = load_frag(BT, ldb, nt0 * 16, kb * 32);
            v16h b1 = load_frag(BT, ldb, nt1 * 16, kb * 32);
            acc0 = __builtin_amdgcn_wmma_f32_16x16x32_f16(
                false, a, false, b0, (short)0, acc0, false, false);
            acc1 = __builtin_amdgcn_wmma_f32_16x16x32_f16(
                false, a, false, b1, (short)0, acc1, false, false);
        }
        epilogue_tile(acc0, mt, nt0, rowb, colw, cf32, cf16, ldc,
                      bias, bias0, scale, relu, storeT);
        epilogue_tile(acc1, mt, nt1, rowb, colw, cf32, cf16, ldc,
                      bias, bias0, scale, relu, storeT);
    }
}

// 4 lanes per row, quad reduction via wave32 shuffles.
__device__ __forceinline__ void layernorm_rows(
    float* __restrict__ h, _Float16* __restrict__ out16,
    const float* __restrict__ g, const float* __restrict__ b)
{
    const int row  = threadIdx.x >> 2;   // 64 rows
    const int part = threadIdx.x & 3;    // 4 x 128 columns
    float s = 0.f, ss = 0.f;
    for (int d = part * 128; d < part * 128 + 128; ++d) {
        float v = h[row * DM + d];
        s += v; ss += v * v;
    }
    s  += __shfl_xor(s, 1, 32);  s  += __shfl_xor(s, 2, 32);
    ss += __shfl_xor(ss, 1, 32); ss += __shfl_xor(ss, 2, 32);
    const float mu   = s * (1.0f / DM);
    const float var  = ss * (1.0f / DM) - mu * mu;
    const float rstd = rsqrtf(var + 1e-6f);
    for (int d = part * 128; d < part * 128 + 128; ++d) {
        float v = (h[row * DM + d] - mu) * rstd * g[d] + b[d];
        h[row * DM + d] = v;
        out16[row * DM + d] = (_Float16)v;
    }
}

// Softmax over 64 cols; writes probs to global atts output and f16 LDS buf.
__device__ __forceinline__ void softmax_rows(
    float* __restrict__ sc, _Float16* __restrict__ p16,
    float* __restrict__ att_out)
{
    const int row  = threadIdx.x >> 2;
    const int part = threadIdx.x & 3;    // 4 x 16 columns
    float m = -1e30f;
    for (int c = part * 16; c < part * 16 + 16; ++c)
        m = fmaxf(m, sc[row * SQ + c]);
    m = fmaxf(m, __shfl_xor(m, 1, 32));
    m = fmaxf(m, __shfl_xor(m, 2, 32));
    float sum = 0.f;
    for (int c = part * 16; c < part * 16 + 16; ++c) {
        float e = __expf(sc[row * SQ + c] - m);
        sc[row * SQ + c] = e; sum += e;
    }
    sum += __shfl_xor(sum, 1, 32);
    sum += __shfl_xor(sum, 2, 32);
    const float inv = 1.0f / sum;
    for (int c = part * 16; c < part * 16 + 16; ++c) {
        float p = sc[row * SQ + c] * inv;
        p16[row * SQ + c] = (_Float16)p;
        att_out[(size_t)row * SQ + c] = p;
    }
}

__device__ __forceinline__ int lower_bound_dev(const int* __restrict__ a,
                                               int n, int v) {
    int lo = 0, hi = n;
    while (lo < hi) { int mid = (lo + hi) >> 1; if (a[mid] < v) lo = mid + 1; else hi = mid; }
    return lo;
}

// ---------------------------------------------------------------------------
// One-time weight prep: f32 [L][K][N] -> f16 transposed [L][N][K] so that
// WMMA B-fragments are two contiguous 16B loads.
// ---------------------------------------------------------------------------
__global__ void convert_transpose_kernel(const float* __restrict__ in,
                                         _Float16* __restrict__ out,
                                         int Lc, int Kd, int Nd)
{
    const long long total = (long long)Lc * Kd * Nd;
    for (long long o = (long long)blockIdx.x * blockDim.x + threadIdx.x;
         o < total; o += (long long)gridDim.x * blockDim.x) {
        const long long per = (long long)Kd * Nd;
        const int l = (int)(o / per);
        const long long rem = o - (long long)l * per;
        const int n = (int)(rem / Kd);
        const int k = (int)(rem % Kd);
        out[o] = (_Float16)in[(long long)l * per + (long long)k * Nd + n];
    }
}

// ---------------------------------------------------------------------------
// Fused encoder: one workgroup (256 threads / 8 waves) per image.
// LDS layout (303,120 B total, < 320 KB WGP LDS):
//   hres  f32 [64][512]  131072  running residual (h -> h1 -> h2)
//   abuf  f16 [64][512]   65536  A-operand staging (h / h1 in f16)
//   fbuf  f16 [64][512]   65536  attn-concat, then FFN hidden block
//   qbuf  f16 [64][64]     8192  Q_h, then softmax probs P_h
//   kbuf  f16 [64][64]     8192  K_h ([seq][d] == B^T for Q.K^T)
//   vbufT f16 [64][64]     8192  V_h transposed ([d][seq] == B^T for P.V)
//   sbuf  f32 [64][64]    16384  attention scores
//   gb    int [2]             8  gather bounds
// ---------------------------------------------------------------------------
#define SMEM_BYTES 303120

__global__ void __launch_bounds__(256)
encoder_kernel(
    const float* __restrict__ x,      const float* __restrict__ W_exp,
    const float* __restrict__ b_exp,
    const float* __restrict__ bq,     const float* __restrict__ bk,
    const float* __restrict__ bv,     const float* __restrict__ bo,
    const float* __restrict__ b1,     const float* __restrict__ b2,
    const float* __restrict__ ln1g,   const float* __restrict__ ln1b,
    const float* __restrict__ ln2g,   const float* __restrict__ ln2b,
    const int*   __restrict__ obj_to_img,
    const _Float16* __restrict__ wq16, const _Float16* __restrict__ wk16,
    const _Float16* __restrict__ wv16, const _Float16* __restrict__ wo16,
    const _Float16* __restrict__ w116, const _Float16* __restrict__ w216,
    float* __restrict__ out)
{
    extern __shared__ char smem_raw[];
    float*    hres  = (float*)    (smem_raw);
    _Float16* abuf  = (_Float16*) (smem_raw + 131072);
    _Float16* fbuf  = (_Float16*) (smem_raw + 131072 + 65536);
    _Float16* qbuf  = (_Float16*) (smem_raw + 262144);
    _Float16* kbuf  = (_Float16*) (smem_raw + 270336);
    _Float16* vbufT = (_Float16*) (smem_raw + 278528);
    float*    sbuf  = (float*)    (smem_raw + 286720);
    int*      gb    = (int*)      (smem_raw + 303104);

    const int img = blockIdx.x;
    const int tid = threadIdx.x;

    // DenseExpander: h[s][d] = x[img][d] * W_exp[s] + b_exp[s]
    for (int idx = tid; idx < SQ * DM; idx += 256) {
        const int s = idx >> 9;      // /512
        const int d = idx & (DM - 1);
        float v = x[(size_t)img * DM + d] * W_exp[s] + b_exp[s];
        hres[idx] = v;
        abuf[idx] = (_Float16)v;
    }
    __syncthreads();

    const float inv_sqrt_dh = 0.125f;   // 1/sqrt(64)

    for (int layer = 0; layer < L_NUM; ++layer) {
        const size_t wOff  = (size_t)layer * DM * DM;
        const size_t w1Off = (size_t)layer * DFF * DM;   // [L][F][D] transposed
        const size_t w2Off = (size_t)layer * DM * DFF;   // [L][D][F] transposed

        // Warm next-layer weights toward L2 (global_prefetch_b8).
        if (layer + 1 < L_NUM) {
            __builtin_prefetch(wq16 + wOff + (size_t)DM * DM + tid * 1024, 0, 1);
            __builtin_prefetch(w116 + w1Off + (size_t)DFF * DM + tid * 1024, 0, 1);
        }

        // ---------------- multi-head attention ----------------
        for (int h = 0; h < NH; ++h) {
            const size_t slice = wOff + (size_t)(h * DH) * DM;  // rows n=h*64..h*64+63
            // Q_h, K_h, V_h : [64, 64] = h(f16) @ W[:, h*64:+64] + b
            wmma_gemm(abuf, DM, wq16 + slice, DM, 4, 16,
                      nullptr, qbuf, DH, bq + layer * DM, h * DH,
                      1.0f, false, false, false);
            wmma_gemm(abuf, DM, wk16 + slice, DM, 4, 16,
                      nullptr, kbuf, DH, bk + layer * DM, h * DH,
                      1.0f, false, false, false);
            wmma_gemm(abuf, DM, wv16 + slice, DM, 4, 16,
                      nullptr, vbufT, DH, bv + layer * DM, h * DH,
                      1.0f, false, false, true);       // store V^T
            __syncthreads();

            // scores = Q K^T * scale  (kbuf [seq][d] is exactly B^T)
            wmma_gemm(qbuf, DH, kbuf, DH, 4, 2,
                      sbuf, nullptr, SQ, nullptr, 0,
                      inv_sqrt_dh, false, false, false);
            __syncthreads();

            // softmax; probs -> qbuf (f16) and atts output (f32)
            float* att_out = out + FLAT_ELEMS +
                ((((size_t)layer * NIMG + img) * NH + h) * ATT_PER_HEAD);
            softmax_rows(sbuf, qbuf, att_out);
            __syncthreads();

            // o_h = P V  -> columns [h*64, h*64+64) of attn-concat
            wmma_gemm(qbuf, DH, vbufT, DH, 4, 2,
                      nullptr, fbuf + h * DH, DM, nullptr, 0,
                      1.0f, false, false, false);
            __syncthreads();
        }

        // O projection, accumulated into residual: hres += concat @ Wo + bo
        wmma_gemm(fbuf, DM, wo16 + wOff, DM, 32, 16,
                  hres, nullptr, DM, bo + layer * DM, 0,
                  1.0f, false, true, false);
        __syncthreads();

        // h1 = LN(h + o); f16 copy into abuf
        layernorm_rows(hres, abuf, ln1g + layer * DM, ln1b + layer * DM);
        __syncthreads();

        // ---------------- FFN, F processed in 4 blocks of 512 ----------------
        for (int kb = 0; kb < 4; ++kb) {
            // fbuf = relu(h1 @ W1[:, kb*512:+512] + b1)
            wmma_gemm(abuf, DM, w116 + w1Off + (size_t)(kb * DM) * DM, DM, 32, 16,
                      nullptr, fbuf, DM, b1 + layer * DFF, kb * DM,
                      1.0f, true, false, false);
            __syncthreads();
            // hres += fbuf @ W2[kb*512:+512, :]  (+ b2 once)
            wmma_gemm(fbuf, DM, w216 + w2Off + (size_t)kb * DM, DFF, 32, 16,
                      hres, nullptr, DM,
                      (kb == 0) ? (b2 + layer * DM) : nullptr, 0,
                      1.0f, false, true, false);
            __syncthreads();
        }

        // h = LN(h1 + f); f16 copy for next layer
        layernorm_rows(hres, abuf, ln2g + layer * DM, ln2b + layer * DM);
        __syncthreads();
    }

    // ---------------- reflat_obj_vecs gather ----------------
    if (tid == 0) {
        gb[0] = lower_bound_dev(obj_to_img, NOBJ, img);
        gb[1] = lower_bound_dev(obj_to_img, NOBJ, img + 1);
    }
    __syncthreads();
    const int start = gb[0], end = gb[1];

#if defined(HAVE_TDM)
    // Final flat rows are hres rows 1..count, contiguous in LDS (row 1 starts
    // at byte 2048). Ship them with the Tensor Data Mover: one 2D descriptor,
    // tile = count*512 f32 contiguous, LDS -> out + start*512. Issued by
    // wave 0 only (TDM ignores EXEC); tracked by TENSORcnt.
    if (tid < 32) {
        int s0  = __builtin_amdgcn_readfirstlane(start);
        int cnt = __builtin_amdgcn_readfirstlane(end - start);
        if (cnt > SQ - 1) cnt = SQ - 1;
        const unsigned n = (unsigned)cnt * DM;             // f32 elements (<= 32256)
        const unsigned long long ga =
            (unsigned long long)(size_t)(out + (size_t)s0 * DM);

        u32x4 g0 = { 1u,                                   // valid user D#
                     2048u,                                // lds_addr = hres row 1
                     (unsigned)ga,                         // global_addr[31:0]
                     (unsigned)((ga >> 32) & 0x1FFFFFFu) | (2u << 30) }; // MSBs|type=2

        i32x8 g1;
        g1[0] = (int)(2u << 16);                           // data_size = 4 bytes
        g1[1] = (int)((n & 0xFFFFu) << 16);                // tensor_dim0 lo (bits 79:48)
        g1[2] = (int)(((n >> 16) & 0xFFFFu) | (1u << 16)); // tensor_dim0 hi | tensor_dim1=1
        g1[3] = (int)((n & 0xFFFFu) << 16);                // tile_dim0 = n
        g1[4] = 1;                                         // tile_dim1 = 1, tile_dim2 = 0
        g1[5] = (int)n;                                    // tensor_dim0_stride lo
        g1[6] = 0;                                         // stride hi | dim1_stride lo
        g1[7] = 0;

        i32x4 z4 = { 0, 0, 0, 0 };
#if __clang_major__ >= 23
        i32x8 z8 = { 0, 0, 0, 0, 0, 0, 0, 0 };
        __builtin_amdgcn_tensor_store_from_lds(g0, g1, z4, z4, z8, 0);
#else
        __builtin_amdgcn_tensor_store_from_lds(g0, g1, z4, z4, 0);
#endif
    }
    __builtin_amdgcn_s_wait_tensorcnt(0);
#else
    for (int j = start; j < end; ++j) {
        int srow = j - start + 1;
        if (srow > SQ - 1) srow = SQ - 1;
        for (int d = tid; d < DM; d += 256)
            out[(size_t)j * DM + d] = hres[srow * DM + d];
    }
#endif
}

// ---------------------------------------------------------------------------
extern "C" void kernel_launch(void* const* d_in, const int* in_sizes, int n_in,
                              void* d_out, int out_size, void* d_ws, size_t ws_size,
                              hipStream_t stream)
{
    (void)in_sizes; (void)n_in; (void)out_size; (void)ws_size;

    const float* x          = (const float*)d_in[0];
    // d_in[1] = masks (zeroed by the module, unused)
    const int*   obj_to_img = (const int*)  d_in[2];
    // d_in[3] = n_imgs scalar (compile-time NIMG)
    const float* W_exp      = (const float*)d_in[4];
    const float* b_exp      = (const float*)d_in[5];
    const float* wq = (const float*)d_in[6];  const float* bq = (const float*)d_in[7];
    const float* wk = (const float*)d_in[8];  const float* bk = (const float*)d_in[9];
    const float* wv = (const float*)d_in[10]; const float* bv = (const float*)d_in[11];
    const float* wo = (const float*)d_in[12]; const float* bo = (const float*)d_in[13];
    const float* w1 = (const float*)d_in[14]; const float* b1 = (const float*)d_in[15];
    const float* w2 = (const float*)d_in[16]; const float* b2 = (const float*)d_in[17];
    const float* ln1g = (const float*)d_in[18]; const float* ln1b = (const float*)d_in[19];
    const float* ln2g = (const float*)d_in[20]; const float* ln2b = (const float*)d_in[21];

    // Workspace: all weights converted to f16, transposed to [L][out][in].
    _Float16* w16 = (_Float16*)d_ws;
    const size_t SZA = (size_t)L_NUM * DM * DM;    // 1,572,864 each
    const size_t SZ1 = (size_t)L_NUM * DM * DFF;   // 6,291,456 each
    _Float16* wq16 = w16;
    _Float16* wk16 = w16 + SZA;
    _Float16* wv16 = w16 + 2 * SZA;
    _Float16* wo16 = w16 + 3 * SZA;
    _Float16* w116 = w16 + 4 * SZA;
    _Float16* w216 = w16 + 4 * SZA + SZ1;

    convert_transpose_kernel<<<2048, 256, 0, stream>>>(wq, wq16, L_NUM, DM, DM);
    convert_transpose_kernel<<<2048, 256, 0, stream>>>(wk, wk16, L_NUM, DM, DM);
    convert_transpose_kernel<<<2048, 256, 0, stream>>>(wv, wv16, L_NUM, DM, DM);
    convert_transpose_kernel<<<2048, 256, 0, stream>>>(wo, wo16, L_NUM, DM, DM);
    convert_transpose_kernel<<<2048, 256, 0, stream>>>(w1, w116, L_NUM, DM, DFF);
    convert_transpose_kernel<<<2048, 256, 0, stream>>>(w2, w216, L_NUM, DFF, DM);

    encoder_kernel<<<NIMG, 256, SMEM_BYTES, stream>>>(
        x, W_exp, b_exp, bq, bk, bv, bo, b1, b2,
        ln1g, ln1b, ln2g, ln2b, obj_to_img,
        wq16, wk16, wv16, wo16, w116, w216,
        (float*)d_out);
}
